// GraphSAGE_Net_377957122121
// MI455X (gfx1250) — matrix operand order repaired and checked by reading, not compiled, not run on hardware
//
#include <hip/hip_runtime.h>

#define N_NODES 100000
#define N_EDGES 1600000
#define DIM 128
#define N_GRAPHS 2000
#define NODES_PER_GRAPH 50
#define BN_EPS 1e-5f

typedef __attribute__((ext_vector_type(16))) __bf16 v16bf;
typedef __attribute__((ext_vector_type(8)))  float v8f;

// ---------------------------------------------------------------- utilities

__global__ void zero_f32(float* __restrict__ p, int n) {
  int i = blockIdx.x * blockDim.x + threadIdx.x;
  if (i < n) p[i] = 0.0f;
}

__device__ __forceinline__ void atomic_add_f32(float* p, float v) {
  // lowers to global_atomic_add_f32 (no return -> STOREcnt path)
  __hip_atomic_fetch_add(p, v, __ATOMIC_RELAXED, __HIP_MEMORY_SCOPE_AGENT);
}

// ------------------------------------------------------- edge scatter (mean)
// one wave per edge; lane handles 4 consecutive features (float4 gather,
// 4 f32 atomic adds). lane 0 optionally bumps the in-degree counter.
__global__ void __launch_bounds__(256)
edge_scatter(const float* __restrict__ feat,
             const int* __restrict__ esrc,
             const int* __restrict__ edst,
             float* __restrict__ agg,
             float* __restrict__ cnt,
             int with_cnt) {
  int e    = blockIdx.x * 8 + (threadIdx.x >> 5);
  int lane = threadIdx.x & 31;
  int s = esrc[e];
  int d = edst[e];
  const float4 v = *(const float4*)(feat + (size_t)s * DIM + lane * 4);
  float* dp = agg + (size_t)d * DIM + lane * 4;
  atomic_add_f32(dp + 0, v.x);
  atomic_add_f32(dp + 1, v.y);
  atomic_add_f32(dp + 2, v.z);
  atomic_add_f32(dp + 3, v.w);
  if (with_cnt && lane == 0) atomic_add_f32(cnt + d, 1.0f);
}

// --------------------------------------------------- WMMA tile loaders (bf16)
// A-matrix 16x32 (MxK), 16-bit layout (ISA 7.12.2):
//   lanes 0-15: row M=lane, K = kBase+0..7 in v0..v3, kBase+16..23 in v4..v7
//   lanes 16-31: row M=lane-16, K groups shifted by +8
__device__ __forceinline__ v16bf load_a(const float* __restrict__ rowp,
                                        int kBase, int lane, float scale) {
  int kb = kBase + ((lane >> 4) << 3);
  v16bf a;
#pragma unroll
  for (int i = 0; i < 8; ++i) a[i]     = (__bf16)(rowp[kb + i]      * scale);
#pragma unroll
  for (int i = 0; i < 8; ++i) a[8 + i] = (__bf16)(rowp[kb + 16 + i] * scale);
  return a;
}

// B-matrix 32x16 (KxN) with B = W^T (out[m,n] = sum_k A[m,k]*W[n,k]):
//   lane holds column n = nBase+(lane&15); lanes 0-15 get K=kBase+0..7,16..23,
//   lanes 16-31 get K=kBase+8..15,24..31 (same K order as load_a).
__device__ __forceinline__ v16bf load_b(const float* __restrict__ W,
                                        int nBase, int kBase, int lane) {
  int n  = nBase + (lane & 15);
  int kb = kBase + ((lane >> 4) << 3);
  const float* p = W + (size_t)n * DIM + kb;
  v16bf b;
#pragma unroll
  for (int i = 0; i < 16; ++i) b[i] = (__bf16)p[(i & 7) + ((i >> 3) << 4)];
  return b;
}

// ------------------------------------------- layer 1: SAGE + bias + BN + ReLU
// h = relu( BN( (agg/deg) @ W1l^T + b1l + x @ W1r^T ) )
__global__ void __launch_bounds__(256)
sage_gemm1(const float* __restrict__ agg, const float* __restrict__ cnt,
           const float* __restrict__ x,
           const float* __restrict__ W1l, const float* __restrict__ b1l,
           const float* __restrict__ W1r,
           const float* __restrict__ gamma, const float* __restrict__ beta,
           const float* __restrict__ rmean, const float* __restrict__ rvar,
           float* __restrict__ h) {
  int gwave = blockIdx.x * 8 + (threadIdx.x >> 5);
  int lane  = threadIdx.x & 31;
  int rowBase = (gwave >> 3) * 16;   // 6250 M tiles
  int nBase   = (gwave & 7) * 16;    // 8 N tiles

  int rowA = rowBase + (lane & 15);
  float invc = 1.0f / fmaxf(cnt[rowA], 1.0f);
  const float* aggRow = agg + (size_t)rowA * DIM;
  const float* xRow   = x   + (size_t)rowA * DIM;

  v8f c = {};
#pragma unroll
  for (int kb = 0; kb < DIM; kb += 32) {
    v16bf a = load_a(aggRow, kb, lane, invc);
    v16bf b = load_b(W1l, nBase, kb, lane);
    c = __builtin_amdgcn_wmma_f32_16x16x32_bf16(false, a, false, b,
                                                (short)0, c, false, false);
  }
#pragma unroll
  for (int kb = 0; kb < DIM; kb += 32) {
    v16bf a = load_a(xRow, kb, lane, 1.0f);
    v16bf b = load_b(W1r, nBase, kb, lane);
    c = __builtin_amdgcn_wmma_f32_16x16x32_bf16(false, a, false, b,
                                                (short)0, c, false, false);
  }

  // epilogue: bias + batchnorm(eval) + relu
  int col = nBase + (lane & 15);
  float sc = gamma[col] * rsqrtf(rvar[col] + BN_EPS);
  float sh = beta[col] + (b1l[col] - rmean[col]) * sc;
  int rtop = rowBase + ((lane >> 4) << 3);   // C/D: vgpr i -> M = i (+8 hi half)
#pragma unroll
  for (int i = 0; i < 8; ++i) {
    float v = fmaxf(c[i] * sc + sh, 0.0f);
    h[(size_t)(rtop + i) * DIM + col] = v;
  }
}

// -------------------------------- layer 2: SAGE + bias + fused mean pooling
// pooled[g] += ((agg/deg) @ W2l^T + b2l + h @ W2r^T)[node] / 50
__global__ void __launch_bounds__(256)
sage_gemm2(const float* __restrict__ agg, const float* __restrict__ cnt,
           const float* __restrict__ h,
           const float* __restrict__ W2l, const float* __restrict__ b2l,
           const float* __restrict__ W2r,
           float* __restrict__ out) {
  int gwave = blockIdx.x * 8 + (threadIdx.x >> 5);
  int lane  = threadIdx.x & 31;
  int rowBase = (gwave >> 3) * 16;
  int nBase   = (gwave & 7) * 16;

  int rowA = rowBase + (lane & 15);
  float invc = 1.0f / fmaxf(cnt[rowA], 1.0f);
  const float* aggRow = agg + (size_t)rowA * DIM;
  const float* hRow   = h   + (size_t)rowA * DIM;

  v8f c = {};
#pragma unroll
  for (int kb = 0; kb < DIM; kb += 32) {
    v16bf a = load_a(aggRow, kb, lane, invc);
    v16bf b = load_b(W2l, nBase, kb, lane);
    c = __builtin_amdgcn_wmma_f32_16x16x32_bf16(false, a, false, b,
                                                (short)0, c, false, false);
  }
#pragma unroll
  for (int kb = 0; kb < DIM; kb += 32) {
    v16bf a = load_a(hRow, kb, lane, 1.0f);
    v16bf b = load_b(W2r, nBase, kb, lane);
    c = __builtin_amdgcn_wmma_f32_16x16x32_bf16(false, a, false, b,
                                                (short)0, c, false, false);
  }

  int col = nBase + (lane & 15);
  float bias = b2l[col];
  int rtop = rowBase + ((lane >> 4) << 3);
#pragma unroll
  for (int i = 0; i < 8; ++i) {
    int row = rtop + i;
    int g = row / NODES_PER_GRAPH;
    atomic_add_f32(out + (size_t)g * DIM + col,
                   (c[i] + bias) * (1.0f / NODES_PER_GRAPH));
  }
}

// ------------------------------------------------------------------ launcher

extern "C" void kernel_launch(void* const* d_in, const int* in_sizes, int n_in,
                              void* d_out, int out_size, void* d_ws, size_t ws_size,
                              hipStream_t stream) {
  const float* x     = (const float*)d_in[0];
  const int*   esrc  = (const int*)d_in[1];
  const int*   edst  = (const int*)d_in[2];
  // d_in[3] = scope (constant NODES_PER_GRAPH, folded in)
  const float* W1l   = (const float*)d_in[4];
  const float* b1l   = (const float*)d_in[5];
  const float* W1r   = (const float*)d_in[6];
  const float* gamma = (const float*)d_in[7];
  const float* beta  = (const float*)d_in[8];
  const float* rmean = (const float*)d_in[9];
  const float* rvar  = (const float*)d_in[10];
  const float* W2l   = (const float*)d_in[11];
  const float* b2l   = (const float*)d_in[12];
  const float* W2r   = (const float*)d_in[13];
  float* out = (float*)d_out;

  float* cnt = (float*)d_ws;                        // [N_NODES]
  float* agg = cnt + N_NODES;                       // [N_NODES, DIM] (reused)
  float* h   = agg + (size_t)N_NODES * DIM;         // [N_NODES, DIM]

  const int nagg = N_NODES * DIM;
  const int gemm_blocks = (N_NODES / 16) * 8 / 8;   // 6250 blocks * 8 waves

  zero_f32<<<(N_NODES + 255) / 256, 256, 0, stream>>>(cnt, N_NODES);
  zero_f32<<<(nagg + 255) / 256, 256, 0, stream>>>(agg, nagg);
  edge_scatter<<<N_EDGES / 8, 256, 0, stream>>>(x, esrc, edst, agg, cnt, 1);
  sage_gemm1<<<gemm_blocks, 256, 0, stream>>>(agg, cnt, x, W1l, b1l, W1r,
                                              gamma, beta, rmean, rvar, h);
  zero_f32<<<(nagg + 255) / 256, 256, 0, stream>>>(agg, nagg);
  edge_scatter<<<N_EDGES / 8, 256, 0, stream>>>(h, esrc, edst, agg, cnt, 0);
  zero_f32<<<(N_GRAPHS * DIM + 255) / 256, 256, 0, stream>>>(out, N_GRAPHS * DIM);
  sage_gemm2<<<gemm_blocks, 256, 0, stream>>>(agg, cnt, h, W2l, b2l, W2r, out);
}
